// CausalSelfAttention_newsim_weight_19645180412430
// MI455X (gfx1250) — matrix operand order repaired and checked by reading, not compiled, not run on hardware
//
#include <hip/hip_runtime.h>
#include <hip/hip_bf16.h>

typedef __bf16 bf16_t;
typedef __attribute__((ext_vector_type(16))) __bf16 v16bf;
typedef __attribute__((ext_vector_type(8)))  __bf16 v8bf;
typedef __attribute__((ext_vector_type(8)))  float  v8f;
typedef __attribute__((ext_vector_type(4)))  float  f32x4;
typedef __attribute__((ext_vector_type(4)))  unsigned int u32x4;
typedef __attribute__((ext_vector_type(8)))  int i32x8;
typedef __attribute__((ext_vector_type(4)))  int i32x4;

#define C_DIM 1024
#define T_DIM 2048
#define H_DIM 16
#define D_DIM 64
#define BH_DIM 32   // B*H
#define BT_DIM 4096 // B*T

__device__ __forceinline__ v8f wmma_bf16(v16bf a, v16bf b, v8f c) {
  // (neg_a, A, neg_b, B, c_mod, C, reuse_a, reuse_b)
  return __builtin_amdgcn_wmma_f32_16x16x32_bf16(false, a, false, b, (short)0, c,
                                                 false, false);
}

// 16-bit A-matrix 16x32: lanes 0-15 hold K = {8h..8h+7, 16+8h..16+8h+7}.
__device__ __forceinline__ v16bf a_frag_bf16(const bf16_t* __restrict__ row, int k0, int h) {
  v8bf lo = *(const v8bf*)(row + k0 + 8 * h);
  v8bf hi = *(const v8bf*)(row + k0 + 16 + 8 * h);
  v16bf a;
#pragma unroll
  for (int j = 0; j < 8; ++j) { a[j] = lo[j]; a[8 + j] = hi[j]; }
  return a;
}

// 16-bit B-matrix 32x16: lane = column n; lane-half h holds K = {8h..8h+7, 16+8h..16+8h+7}.
__device__ __forceinline__ v16bf b_frag_bf16(const bf16_t* __restrict__ row, int k0, int h) {
  v8bf lo = *(const v8bf*)(row + k0 + 8 * h);
  v8bf hi = *(const v8bf*)(row + k0 + 16 + 8 * h);
  v16bf b;
#pragma unroll
  for (int j = 0; j < 8; ++j) { b[j] = lo[j]; b[8 + j] = hi[j]; }
  return b;
}

__device__ __forceinline__ float half_max(float v) {
#pragma unroll
  for (int off = 8; off >= 1; off >>= 1) v = fmaxf(v, __shfl_xor(v, off, 32));
  return v;
}
__device__ __forceinline__ float half_sum(float v) {
#pragma unroll
  for (int off = 8; off >= 1; off >>= 1) v += __shfl_xor(v, off, 32);
  return v;
}

// ---- Tensor Data Mover: 2D bf16 tile (tile_k x tile_rows) -> LDS (ISA ch.8 D#)
__device__ __forceinline__ void tdm_load_b_tile(unsigned lds_addr,
                                                const bf16_t* __restrict__ gptr) {
  const unsigned long long ga = (unsigned long long)(uintptr_t)gptr;
  u32x4 g0;
  g0[0] = 1u;                                              // count=1, user D#
  g0[1] = lds_addr;                                        // LDS byte address
  g0[2] = (unsigned)(ga & 0xFFFFFFFFull);                  // global_addr[31:0]
  g0[3] = (unsigned)((ga >> 32) & 0x01FFFFFFull) | (2u << 30);  // [56:32] | type=2
  i32x8 g1;
  const unsigned dim0 = C_DIM;      // tensor row length (elements)
  const unsigned dim1 = C_DIM;      // tensor rows
  const unsigned tile_k = 32;       // tile_dim0
  const unsigned tile_rows = 64;    // tile_dim1
  const unsigned stride0 = C_DIM;   // tensor_dim0_stride (elements)
  g1[0] = (int)(1u << 16);                                  // data_size=1 (2 bytes)
  g1[1] = (int)((dim0 & 0xFFFFu) << 16);                    // tensor_dim0[15:0]
  g1[2] = (int)(((dim0 >> 16) & 0xFFFFu) | ((dim1 & 0xFFFFu) << 16));
  g1[3] = (int)(((dim1 >> 16) & 0xFFFFu) | (tile_k << 16)); // tile_dim0
  g1[4] = (int)tile_rows;                                   // tile_dim1 (tile_dim2=0)
  g1[5] = (int)stride0;                                     // stride0[31:0]
  g1[6] = 0;                                                // stride0[47:32]|stride1 lo
  g1[7] = 0;
  i32x4 z4 = {0, 0, 0, 0};
  i32x8 z8 = {0, 0, 0, 0, 0, 0, 0, 0};
  // 6-arg variant (clang-23 / therock headers): (g0, g1, g2, g3, g4, cpol)
  __builtin_amdgcn_tensor_load_to_lds(g0, g1, z4, z4, z8, 0);
}

// ---- shared GEMM mainloop: acc[4] (16M x 64N), B tile TDM-staged + dbl-buffered
__device__ __forceinline__ void gemm_mainloop(
    const bf16_t* __restrict__ arow,     // lane's A row (length C_DIM, bf16)
    const bf16_t* __restrict__ Bbf,      // [C_DIM][C_DIM] bf16, row = output col
    int n0, int wave, int lm, int h,
    bf16_t (*tileB)[64 * 32], v8f acc[4]) {
  if (wave == 0) {
    tdm_load_b_tile((unsigned)(uintptr_t)&tileB[0][0], Bbf + (size_t)n0 * C_DIM);
  }
  for (int k0 = 0; k0 < C_DIM; k0 += 32) {
    const int cur = (k0 >> 5) & 1;
    if (wave == 0) __builtin_amdgcn_s_wait_tensorcnt((short)0);  // cur tile landed
    __syncthreads();   // cur visible to all; all done with buffer cur^1
    if (wave == 0 && (k0 + 32) < C_DIM) {
      tdm_load_b_tile((unsigned)(uintptr_t)&tileB[cur ^ 1][0],
                      Bbf + (size_t)n0 * C_DIM + (k0 + 32));
    }
    const v16bf a = a_frag_bf16(arow, k0, h);
    const bf16_t* bt = &tileB[cur][0];
#pragma unroll
    for (int j = 0; j < 4; ++j) {
      acc[j] = wmma_bf16(a, b_frag_bf16(bt + (j * 16 + lm) * 32, 0, h), acc[j]);
    }
  }
}

// ---------------- Kernel 0: fp32 -> bf16 bulk convert (8 elements/thread)
__global__ __launch_bounds__(256) void cvt_bf16_kernel(
    const float* __restrict__ src, bf16_t* __restrict__ dst, int n8) {
  const int i = blockIdx.x * 256 + threadIdx.x;
  if (i < n8) {
    const f32x4* p = (const f32x4*)(src + (size_t)i * 8);
    f32x4 a = p[0], b = p[1];
    v8bf o;
#pragma unroll
    for (int j = 0; j < 4; ++j) { o[j] = (__bf16)a[j]; o[4 + j] = (__bf16)b[j]; }
    *(v8bf*)(dst + (size_t)i * 8) = o;
  }
}

// ---------------- Kernel 1: qkv = x @ Wv^T + bv ; emit q/k (bf16) + v^T (bf16)
__global__ __launch_bounds__(256) void qkv_proj_kernel(
    const bf16_t* __restrict__ xbf, const bf16_t* __restrict__ Wvbf,
    const float* __restrict__ weight, const float* __restrict__ bv,
    const float* __restrict__ v_scale,
    bf16_t* __restrict__ qk, bf16_t* __restrict__ vT) {
  __shared__ bf16_t tileB[2][64 * 32];   // 8 KB double-buffered B tile
  const int lane = threadIdx.x & 31;
  const int wave = threadIdx.x >> 5;
  const int lm = lane & 15;
  const int h  = lane >> 4;
  const int n0  = (blockIdx.x & 15) * 64;        // 16 N-blocks
  const int m0w = (blockIdx.x >> 4) * 128 + wave * 16;  // 32 M-blocks x 8 waves

  v8f acc[4];
  v8f vzero = {};
#pragma unroll
  for (int j = 0; j < 4; ++j) acc[j] = vzero;

  gemm_mainloop(xbf + (size_t)(m0w + lm) * C_DIM, Wvbf, n0, wave, lm, h, tileB, acc);

  const float vs = v_scale[0];
#pragma unroll
  for (int j = 0; j < 4; ++j) {
    const int col  = n0 + j * 16 + lm;
    const int head = col >> 6;
    const int dd   = col & 63;
    const float bias = bv[col];
#pragma unroll
    for (int r = 0; r < 8; ++r) {
      const int mrow = m0w + r + 8 * h;     // C-layout: lanes16-31 hold M=8+r
      const int bidx = mrow >> 11;
      const int t    = mrow & (T_DIM - 1);
      const int bh   = bidx * H_DIM + head;
      const float val = acc[j][r] + bias;
      qk[((size_t)bh * T_DIM + t) * D_DIM + dd] = (bf16_t)val;
      const float wv = weight[mrow] * vs;
      vT[((size_t)bh * D_DIM + dd) * T_DIM + t] = (bf16_t)(val * wv);
    }
  }
}

// ---------------- Kernel 2: causal flash attention (one wave = 16 query rows)
__global__ __launch_bounds__(256) void flash_attn_kernel(
    const bf16_t* __restrict__ qk, const bf16_t* __restrict__ vT,
    const float* __restrict__ weight, const float* __restrict__ att_scale,
    bf16_t* __restrict__ y) {
  __shared__ bf16_t Plds[8][16 * 32];   // per-wave 16x32 P staging (8 KB)

  const int lane = threadIdx.x & 31;
  const int wave = threadIdx.x >> 5;
  const int lm = lane & 15;
  const int h  = lane >> 4;
  const int bh   = blockIdx.x >> 4;
  const int qblk = blockIdx.x & 15;
  const int t0   = qblk * 128 + wave * 16;
  const int bidx = bh >> 4;

  const bf16_t* qbase = qk + ((size_t)bh * T_DIM + (t0 + lm)) * D_DIM;
  const v16bf a0 = a_frag_bf16(qbase, 0, h);
  const v16bf a1 = a_frag_bf16(qbase, 32, h);

  const float ascale = att_scale[0] * 0.125f;   // 1/sqrt(64)
  const float NEG_INF = -__builtin_inff();

  float rowM[8], rowL[8];
  v8f acc[4];
  v8f vzero = {};
#pragma unroll
  for (int r = 0; r < 8; ++r) { rowM[r] = NEG_INF; rowL[r] = 0.0f; }
#pragma unroll
  for (int i = 0; i < 4; ++i) acc[i] = vzero;

  const int smax = t0 + 15;
  bf16_t* pl = &Plds[wave][0];

  for (int s0 = 0; s0 <= smax; s0 += 32) {
    float sc[2][8];
#pragma unroll
    for (int nb = 0; nb < 2; ++nb) {
      const int scol = s0 + nb * 16 + lm;
      const bf16_t* krow = qk + ((size_t)bh * T_DIM + scol) * D_DIM;
      v8f s = wmma_bf16(a0, b_frag_bf16(krow, 0, h), vzero);
      s = wmma_bf16(a1, b_frag_bf16(krow, 32, h), s);
      const float wk = weight[bidx * T_DIM + scol] * ascale;
#pragma unroll
      for (int r = 0; r < 8; ++r) {
        const int trow = t0 + r + 8 * h;
        sc[nb][r] = (scol <= trow) ? s[r] * wk : NEG_INF;
      }
    }
#pragma unroll
    for (int r = 0; r < 8; ++r) {
      float mx = half_max(fmaxf(sc[0][r], sc[1][r]));
      const float nm    = fmaxf(rowM[r], mx);
      const float alpha = __expf(rowM[r] - nm);
      const float p0 = __expf(sc[0][r] - nm);
      const float p1 = __expf(sc[1][r] - nm);
      rowL[r] = rowL[r] * alpha + half_sum(p0 + p1);
      rowM[r] = nm;
#pragma unroll
      for (int i = 0; i < 4; ++i) acc[i][r] *= alpha;
      const int prow = r + 8 * h;
      pl[prow * 32 + lm]      = (bf16_t)p0;
      pl[prow * 32 + 16 + lm] = (bf16_t)p1;
    }
    __asm__ volatile("s_wait_dscnt 0" ::: "memory");
    const v16bf pa = a_frag_bf16(pl + lm * 32, 0, h);
#pragma unroll
    for (int i = 0; i < 4; ++i) {
      const bf16_t* vrow = vT + ((size_t)bh * D_DIM + i * 16 + lm) * T_DIM + s0;
      acc[i] = wmma_bf16(pa, b_frag_bf16(vrow, 0, h), acc[i]);
    }
    if (s0 + 32 <= smax) {
      __builtin_prefetch(qk + ((size_t)bh * T_DIM + (s0 + 32 + lm)) * D_DIM, 0, 0);
    }
  }

  const int headc = (bh & 15) * D_DIM;
#pragma unroll
  for (int i = 0; i < 4; ++i) {
    const int colc = headc + i * 16 + lm;
#pragma unroll
    for (int r = 0; r < 8; ++r) {
      const int trow = t0 + r + 8 * h;
      y[((size_t)(bidx * T_DIM + trow)) * C_DIM + colc] =
          (bf16_t)(acc[i][r] / rowL[r]);
    }
  }
}

// ---------------- Kernel 3: out = y @ Wp^T + bp (fp32 out)
__global__ __launch_bounds__(256) void out_proj_kernel(
    const bf16_t* __restrict__ y, const bf16_t* __restrict__ Wpbf,
    const float* __restrict__ bp, float* __restrict__ out) {
  __shared__ bf16_t tileB[2][64 * 32];
  const int lane = threadIdx.x & 31;
  const int wave = threadIdx.x >> 5;
  const int lm = lane & 15;
  const int h  = lane >> 4;
  const int n0  = (blockIdx.x & 15) * 64;
  const int m0w = (blockIdx.x >> 4) * 128 + wave * 16;

  v8f acc[4];
  v8f vzero = {};
#pragma unroll
  for (int j = 0; j < 4; ++j) acc[j] = vzero;

  gemm_mainloop(y + (size_t)(m0w + lm) * C_DIM, Wpbf, n0, wave, lm, h, tileB, acc);

#pragma unroll
  for (int j = 0; j < 4; ++j) {
    const int col = n0 + j * 16 + lm;
    const float bias = bp[col];
#pragma unroll
    for (int r = 0; r < 8; ++r) {
      out[(size_t)(m0w + r + 8 * h) * C_DIM + col] = acc[j][r] + bias;
    }
  }
}

extern "C" void kernel_launch(void* const* d_in, const int* in_sizes, int n_in,
                              void* d_out, int out_size, void* d_ws, size_t ws_size,
                              hipStream_t stream) {
  (void)in_sizes; (void)n_in; (void)out_size; (void)ws_size;
  const float* x         = (const float*)d_in[0];
  const float* weight    = (const float*)d_in[1];
  const float* Wv        = (const float*)d_in[2];
  const float* bv        = (const float*)d_in[3];
  const float* Wp        = (const float*)d_in[4];
  const float* bp        = (const float*)d_in[5];
  const float* v_scale   = (const float*)d_in[6];
  const float* att_scale = (const float*)d_in[7];
  float* out = (float*)d_out;

  const size_t headElems = (size_t)BH_DIM * T_DIM * D_DIM;   // 4,194,304
  const size_t wElems    = (size_t)C_DIM * C_DIM;            // 1,048,576
  bf16_t* qk   = (bf16_t*)d_ws;
  bf16_t* vT   = qk + headElems;
  bf16_t* y    = vT + headElems;
  bf16_t* xbf  = y + headElems;
  bf16_t* wvbf = xbf + (size_t)BT_DIM * C_DIM;
  bf16_t* wpbf = wvbf + wElems;

  // one-shot fp32 -> bf16 conversions (removes per-tile cvt + halves GEMM bytes)
  cvt_bf16_kernel<<<(BT_DIM * C_DIM / 8) / 256, 256, 0, stream>>>(x, xbf,
                                                                  BT_DIM * C_DIM / 8);
  cvt_bf16_kernel<<<(int)(wElems / 8) / 256, 256, 0, stream>>>(Wv, wvbf,
                                                               (int)(wElems / 8));
  cvt_bf16_kernel<<<(int)(wElems / 8) / 256, 256, 0, stream>>>(Wp, wpbf,
                                                               (int)(wElems / 8));

  // 32 M-blocks (128 rows) x 16 N-blocks (64 cols) = 512 blocks
  qkv_proj_kernel<<<512, 256, 0, stream>>>(xbf, wvbf, weight, bv, v_scale, qk, vT);
  // 32 (b,h) x 16 q-blocks of 128 rows = 512 blocks
  flash_attn_kernel<<<512, 256, 0, stream>>>(qk, vT, weight, att_scale, y);
  out_proj_kernel<<<512, 256, 0, stream>>>(y, wpbf, bp, out);
}
